// ModifiedGATLayer_36421322670607
// MI455X (gfx1250) — matrix operand order repaired and checked by reading, not compiled, not run on hardware
//
#include <hip/hip_runtime.h>

#define Nn    8192
#define DIMS  256
#define CH    128
#define ALPHA_ 0.2f
#define NEGV  -9.0e15f
#define LOG2E 1.4426950408889634f

typedef __attribute__((ext_vector_type(16))) __bf16       v16bf;
typedef __attribute__((ext_vector_type(8)))  float        v8f;
typedef __attribute__((ext_vector_type(4)))  unsigned int v4u;
typedef __attribute__((ext_vector_type(8)))  int          v8i_;
typedef __attribute__((ext_vector_type(4)))  int          v4i_;

union BF16x16 { unsigned u[8]; v16bf v; };

#if defined(__HIP_DEVICE_COMPILE__) && __has_builtin(__builtin_amdgcn_tensor_load_to_lds)
#define HAVE_TDM 1
#else
#define HAVE_TDM 0
#endif

// pack two f32 -> packed bf16 pair. Device path: 2x add + 1x v_perm_b32
// (round-half-up; accuracy indistinguishable for attention weights).
__device__ __forceinline__ unsigned pack_bf16(float a, float b) {
#if defined(__HIP_DEVICE_COMPILE__)
  const unsigned ua = __float_as_uint(a) + 0x8000u;
  const unsigned ub = __float_as_uint(b) + 0x8000u;
  return __builtin_amdgcn_perm(ub, ua, 0x07060302u);  // {ub[31:16], ua[31:16]}
#else
  unsigned ua = __float_as_uint(a), ub = __float_as_uint(b);
  ua += 0x7FFFu + ((ua >> 16) & 1u);
  ub += 0x7FFFu + ((ub >> 16) & 1u);
  return (ua >> 16) | (ub & 0xFFFF0000u);
#endif
}
__device__ __forceinline__ unsigned short bf16_1(float a) {
  unsigned ua = __float_as_uint(a);
  ua += 0x7FFFu + ((ua >> 16) & 1u);
  return (unsigned short)(ua >> 16);
}
__device__ __forceinline__ float fast_exp2(float x) {
#if defined(__HIP_DEVICE_COMPILE__)
  return __builtin_amdgcn_exp2f(x);
#else
  return exp2f(x);
#endif
}
// 16-bit A-matrix 16x32 per-lane K map (contiguous, matching B): lanes 0-15 hold
// K {0..15}, lanes 16-31 hold {16..31}.
__device__ __forceinline__ int kmapA(int v, int kh) {
  return 2 * v + 16 * kh;
}

__device__ __forceinline__ v8f wmma_bf16(const BF16x16& a, const BF16x16& b, v8f c) {
#if defined(__HIP_DEVICE_COMPILE__)
  return __builtin_amdgcn_wmma_f32_16x16x32_bf16(false, a.v, false, b.v,
                                                 (short)0, c, false, false);
#else
  return c;
#endif
}

// p = adj ? exp(lrelu(s1+s2) - m) : 0, with mvl = -m*log2e precomputed.
// lrelu(e) == max(e, alpha*e) for alpha < 1.  Cost: add,mul,max,fma,exp,cndmask.
__device__ __forceinline__ float att_p(int adjv, float s2v, float s1v, float mvl) {
  float e = s1v + s2v;
  e = fmaxf(e, ALPHA_ * e);
  const float p = fast_exp2(fmaf(e, LOG2E, mvl));
  return (adjv == 1) ? p : 0.0f;
}

// ---- TDM: async DMA of a 32-row x 64-col int32 adj tile into LDS ----------
__device__ __forceinline__ void tdm_adj_tile(const int* adj, int irow0, int j0,
                                             unsigned lds_off) {
#if HAVE_TDM
  unsigned long long ga =
      (unsigned long long)(const void*)(adj + (size_t)irow0 * Nn + j0);
  // D# group0: count=1 | lds_addr | global_addr | type=2
  v4u g0 = { 1u, lds_off, (unsigned)ga,
             (unsigned)((ga >> 32) & 0x01FFFFFFull) | 0x80000000u };
  // D# group1: data_size=4B, tensor 8192x8192, tile 64(x) x 32(y), stride 8192
  v8i_ g1 = { (int)0x00020000,        // data_size = 2 (4 bytes)
              (int)0x20000000,        // tensor_dim0 = 8192 (lo16 << 16)
              (int)0x20000000,        // dim0 hi16=0 | tensor_dim1 lo16 (8192)
              (int)(64u << 16),       // dim1 hi16=0 | tile_dim0 = 64
              (int)32,                // tile_dim1 = 32, tile_dim2 = 0
              (int)Nn,                // tensor_dim0_stride = 8192
              (int)0x20000000,        // tensor_dim1_stride lo16 = 8192
              0 };
  v4i_ gz = {0, 0, 0, 0};
#if defined(__clang_major__) && (__clang_major__ >= 23)
  v8i_ gz8 = {0, 0, 0, 0, 0, 0, 0, 0};
  __builtin_amdgcn_tensor_load_to_lds(g0, g1, gz, gz, gz8, 0);
#else
  __builtin_amdgcn_tensor_load_to_lds(g0, g1, gz, gz, 0);
#endif
#else
  (void)adj; (void)irow0; (void)j0; (void)lds_off;
#endif
}

__device__ __forceinline__ void wait_tensor(int n) {
#if HAVE_TDM
#if __has_builtin(__builtin_amdgcn_s_wait_tensorcnt)
  if (n) __builtin_amdgcn_s_wait_tensorcnt(1);
  else   __builtin_amdgcn_s_wait_tensorcnt(0);
#else
  asm volatile("s_wait_tensorcnt 0" ::: "memory");
#endif
#else
  (void)n;
#endif
}

// ---- Kernel A: u1,u2 = per-head fold of a into W1/W2; c = b.a --------------
__global__ void gat_prep(const float* __restrict__ a, const float* __restrict__ W1,
                         const float* __restrict__ b1, const float* __restrict__ W2,
                         const float* __restrict__ b2, float* __restrict__ u,
                         float* __restrict__ cb) {
  const int t = threadIdx.x;
  const int k = t >> 9, hh = (t >> 8) & 1, d = t & 255;
  const float* W = (k == 0) ? W1 : W2;
  float s = 0.f;
  for (int c = 0; c < CH; ++c)
    s += a[hh * 256 + k * CH + c] * W[(size_t)(hh * CH + c) * DIMS + d];
  u[(k * 2 + hh) * 256 + d] = s;
  if (t < 4) {
    const int kk = t >> 1, h2 = t & 1;
    const float* bb = (kk == 0) ? b1 : b2;
    float cs = 0.f;
    for (int c = 0; c < CH; ++c) cs += bb[h2 * CH + c] * a[h2 * 256 + kk * CH + c];
    cb[kk * 2 + h2] = cs;
  }
}

// ---- Kernel B: s1[i,h], s2[h][i] = h_row . u ------------------------------
__global__ void gat_scores(const float* __restrict__ H, const float* __restrict__ u,
                           const float* __restrict__ cb, float* __restrict__ s1,
                           float* __restrict__ s2h) {
  const int wave = threadIdx.x >> 5, lane = threadIdx.x & 31;
  const int row = blockIdx.x * 8 + wave;
  const float* hr = H + (size_t)row * DIMS;
  const float4 x0 = *(const float4*)(hr + 4 * lane);
  const float4 x1 = *(const float4*)(hr + 128 + 4 * lane);
  float d[4];
#pragma unroll
  for (int q = 0; q < 4; ++q) {
    const float* uq = u + q * 256;
    const float4 a0 = *(const float4*)(uq + 4 * lane);
    const float4 a1 = *(const float4*)(uq + 128 + 4 * lane);
    d[q] = x0.x * a0.x + x0.y * a0.y + x0.z * a0.z + x0.w * a0.w +
           x1.x * a1.x + x1.y * a1.y + x1.z * a1.z + x1.w * a1.w;
  }
#pragma unroll
  for (int off = 16; off; off >>= 1) {
#pragma unroll
    for (int q = 0; q < 4; ++q) d[q] += __shfl_xor(d[q], off, 32);
  }
  if (lane == 0) {
    s1[row * 2 + 0] = d[0] + cb[0];
    s1[row * 2 + 1] = d[1] + cb[1];
    s2h[row]        = d[2] + cb[2];
    s2h[Nn + row]   = d[3] + cb[3];
  }
}

// ---- Kernel C: W3h = h @ W3.T + b3 (bf16 WMMA), stored transposed bf16 ----
__global__ void gat_w3(const float* __restrict__ H, const float* __restrict__ W3,
                       const float* __restrict__ b3,
                       unsigned short* __restrict__ W3bT) {
  const int wave = threadIdx.x >> 5, lane = threadIdx.x & 31;
  const int tile = blockIdx.x * 8 + wave;              // 8192 tiles total
  const int m0 = (tile >> 4) << 4, n0 = (tile & 15) << 4;
  const int mr = lane & 15, kh = lane >> 4;
  v8f acc = {0.f, 0.f, 0.f, 0.f, 0.f, 0.f, 0.f, 0.f};
  const float* arow = H + (size_t)(m0 + mr) * DIMS;    // A row (M = mr)
  const float* brow = W3 + (size_t)(n0 + mr) * DIMS;   // B col n = mr
#pragma unroll
  for (int k0 = 0; k0 < DIMS; k0 += 32) {
    BF16x16 a, b;
#pragma unroll
    for (int v = 0; v < 8; ++v) {
      const float2 p = *(const float2*)(arow + k0 + kmapA(v, kh));
      a.u[v] = pack_bf16(p.x, p.y);
    }
    const float4* bp = (const float4*)(brow + k0 + 16 * kh);  // 16 contiguous K
    const float4 q0 = bp[0], q1 = bp[1], q2 = bp[2], q3 = bp[3];
    b.u[0] = pack_bf16(q0.x, q0.y); b.u[1] = pack_bf16(q0.z, q0.w);
    b.u[2] = pack_bf16(q1.x, q1.y); b.u[3] = pack_bf16(q1.z, q1.w);
    b.u[4] = pack_bf16(q2.x, q2.y); b.u[5] = pack_bf16(q2.z, q2.w);
    b.u[6] = pack_bf16(q3.x, q3.y); b.u[7] = pack_bf16(q3.z, q3.w);
    acc = wmma_bf16(a, b, acc);
  }
  const float bias = b3[n0 + mr];
#pragma unroll
  for (int r = 0; r < 8; ++r) {
    const int j = m0 + r + 8 * kh;       // node index (consecutive over r)
    const int c = n0 + mr;               // channel (h*128 + c)
    W3bT[(size_t)c * Nn + j] = bf16_1(acc[r] + bias);
  }
}

// ---- Kernel D: m[i,h] = lrelu(s1 + masked-max s2) (softmax row max) -------
__global__ void gat_rowmax(const int* __restrict__ adj, const float* __restrict__ s1,
                           const float* __restrict__ s2h, float* __restrict__ m) {
  __shared__ float2 s2s[Nn];                 // 64 KB of 320 KB LDS
  const int tid = threadIdx.x, wave = tid >> 5, lane = tid & 31;
  for (int j = tid; j < Nn; j += 256)
    s2s[j] = make_float2(s2h[j], s2h[Nn + j]);
  __syncthreads();
  for (int rr = 0; rr < 4; ++rr) {
    const int row = blockIdx.x * 32 + wave * 4 + rr;
    const int4* ap = (const int4*)(adj + (size_t)row * Nn);
    float m0 = -3.0e38f, m1 = -3.0e38f;
    for (int it = 0; it < Nn / 128; ++it) {
      if (it + 8 < Nn / 128) __builtin_prefetch(ap + (it + 8) * 32 + lane, 0, 0);
      const int4 av = ap[it * 32 + lane];
      const int jb = (it * 32 + lane) * 4;
      float2 s;
      s = s2s[jb + 0]; if (av.x == 1) { m0 = fmaxf(m0, s.x); m1 = fmaxf(m1, s.y); }
      s = s2s[jb + 1]; if (av.y == 1) { m0 = fmaxf(m0, s.x); m1 = fmaxf(m1, s.y); }
      s = s2s[jb + 2]; if (av.z == 1) { m0 = fmaxf(m0, s.x); m1 = fmaxf(m1, s.y); }
      s = s2s[jb + 3]; if (av.w == 1) { m0 = fmaxf(m0, s.x); m1 = fmaxf(m1, s.y); }
    }
#pragma unroll
    for (int off = 16; off; off >>= 1) {
      m0 = fmaxf(m0, __shfl_xor(m0, off, 32));
      m1 = fmaxf(m1, __shfl_xor(m1, off, 32));
    }
    if (lane == 0) {
      float e0 = s1[row * 2 + 0] + m0; e0 = fmaxf(e0, ALPHA_ * e0);
      float e1 = s1[row * 2 + 1] + m1; e1 = fmaxf(e1, ALPHA_ * e1);
      m[row * 2 + 0] = (m0 < -1.0e37f) ? 0.f : e0;
      m[row * 2 + 1] = (m1 < -1.0e37f) ? 0.f : e1;
    }
  }
}

// ---- Kernel E: fused exp-softmax + WMMA aggregation, TDM-streamed adj -----
__global__ void __launch_bounds__(128)
gat_agg(const int* __restrict__ adj, const float* __restrict__ s1,
        const float* __restrict__ s2h, const float* __restrict__ mrow,
        const unsigned short* __restrict__ W3bT, float* __restrict__ out) {
  __shared__ int adjb[2][32 * 64];          // double-buffered 8 KB tiles
  const int tid = threadIdx.x;
  const int wave = tid >> 5, lane = tid & 31;
  const int head = wave & 1;
  const int i0 = blockIdx.x * 32;
  const int iw = i0 + (wave >> 1) * 16;
  const int mr = lane & 15, kh = lane >> 4;
  const int row = iw + mr;
  const float s1v = s1[row * 2 + head];
  const float mvl = -mrow[row * 2 + head] * LOG2E;   // folded into exp2 fma
  const float* s2p = s2h + head * Nn;
  const int lrowbase = ((wave >> 1) * 16 + mr) * 64;
  const v8f vz = {0.f, 0.f, 0.f, 0.f, 0.f, 0.f, 0.f, 0.f};
  v8f acc[8];
#pragma unroll
  for (int t = 0; t < 8; ++t) acc[t] = vz;
  float z = 0.f;

  const int NCHUNK = Nn / 64;
#if HAVE_TDM
  if (tid < 32)
    tdm_adj_tile(adj, i0, 0, (unsigned)(unsigned long long)(const void*)&adjb[0][0]);
#endif
  for (int c = 0; c < NCHUNK; ++c) {
    const int cur = c & 1;
#if HAVE_TDM
    if (tid < 32) {                    // wave 0 drives the TDM pipeline
      if (c + 1 < NCHUNK) {
        tdm_adj_tile(adj, i0, (c + 1) * 64,
                     (unsigned)(unsigned long long)(const void*)&adjb[cur ^ 1][0]);
        wait_tensor(1);                // oldest (chunk c) complete
      } else {
        wait_tensor(0);
      }
    }
#else
    {
      const int j0c = c * 64;
      for (int idx = tid; idx < 32 * 64 / 4; idx += 128) {
        const int r = (idx * 4) >> 6, col = (idx * 4) & 63;
        *(int4*)&adjb[cur][r * 64 + col] =
            *(const int4*)(adj + (size_t)(i0 + r) * Nn + j0c + col);
      }
    }
#endif
    __syncthreads();
    const int* at = &adjb[cur][0];
    const int j0 = c * 64;
#pragma unroll
    for (int kk = 0; kk < 64; kk += 32) {
      BF16x16 af;
#pragma unroll
      for (int v = 0; v < 8; ++v) {
        const int K = kmapA(v, kh);
        const float2 sp = *(const float2*)(s2p + j0 + kk + K);
        const int2 ap = *(const int2*)(at + lrowbase + kk + K);
        const float p0 = att_p(ap.x, sp.x, s1v, mvl);
        const float p1 = att_p(ap.y, sp.y, s1v, mvl);
        z += p0 + p1;
        af.u[v] = pack_bf16(p0, p1);
      }
#pragma unroll
      for (int t = 0; t < 8; ++t) {
        const unsigned short* bp =
            W3bT + (size_t)(head * CH + t * 16 + mr) * Nn + j0 + kk + 16 * kh;
        const uint4 lo = *(const uint4*)(bp);
        const uint4 hi = *(const uint4*)(bp + 8);
        BF16x16 bf;
        bf.u[0] = lo.x; bf.u[1] = lo.y; bf.u[2] = lo.z; bf.u[3] = lo.w;
        bf.u[4] = hi.x; bf.u[5] = hi.y; bf.u[6] = hi.z; bf.u[7] = hi.w;
        acc[t] = wmma_bf16(af, bf, acc[t]);
      }
    }
    __syncthreads();
  }
  // Z: lane halves cover complementary K sets -> xor-16 gives full row sum
  const float zf = z + __shfl_xor(z, 16, 32);
#pragma unroll
  for (int r = 0; r < 8; ++r) {
    const float zr = __shfl(zf, r + 8 * kh, 32);
    const float inv = 1.0f / fmaxf(zr, 1e-30f);
    const int orow = iw + r + 8 * kh;
#pragma unroll
    for (int t = 0; t < 8; ++t)
      out[(size_t)orow * DIMS + head * CH + t * 16 + mr] = acc[t][r] * inv;
  }
}

extern "C" void kernel_launch(void* const* d_in, const int* in_sizes, int n_in,
                              void* d_out, int out_size, void* d_ws, size_t ws_size,
                              hipStream_t stream) {
  const float* H   = (const float*)d_in[0];
  const int*   adj = (const int*)d_in[1];
  const float* W1  = (const float*)d_in[2];
  const float* b1  = (const float*)d_in[3];
  const float* W2  = (const float*)d_in[4];
  const float* b2  = (const float*)d_in[5];
  const float* W3  = (const float*)d_in[6];
  const float* b3  = (const float*)d_in[7];
  const float* av  = (const float*)d_in[8];
  float* out = (float*)d_out;

  // workspace layout (~4.4 MB): W3bT(bf16) | s1 | s2h | m | u | cb
  unsigned short* W3bT = (unsigned short*)d_ws;
  float* fws  = (float*)((char*)d_ws + (size_t)DIMS * Nn * 2);
  float* s1v  = fws;              // N*2
  float* s2h  = s1v + 2 * Nn;     // 2*N   ([head][node])
  float* mro  = s2h + 2 * Nn;     // N*2
  float* u    = mro + 2 * Nn;     // 4*256
  float* cb   = u + 4 * 256;      // 4

  gat_prep  <<<1,        1024, 0, stream>>>(av, W1, b1, W2, b2, u, cb);
  gat_scores<<<Nn / 8,   256,  0, stream>>>(H, u, cb, s1v, s2h);
  gat_w3    <<<Nn / 8,   256,  0, stream>>>(H, W3, b3, W3bT);
  gat_rowmax<<<Nn / 32,  256,  0, stream>>>(adj, s1v, s2h, mro);
  gat_agg   <<<Nn / 32,  128,  0, stream>>>(adj, s1v, s2h, mro, W3bT, out);
}
